// Seq2SeqStandardRNN_15942918602975
// MI455X (gfx1250) — matrix operand order repaired and checked by reading, not compiled, not run on hardware
//
#include <hip/hip_runtime.h>

// ---------------------------------------------------------------------------
// Seq2Seq 2-layer RNN (B=64, T=10, D_IN=4096, HID=2048, PRED=10) for gfx1250.
// Strategy: convert weights to bf16 once (fits in 192MB L2), run every GEMM
// as NT bf16 WMMA (v_wmma_f32_16x16x32_bf16) with f32 accumulation.
// ---------------------------------------------------------------------------

typedef __attribute__((ext_vector_type(16))) __bf16 v16bf;
typedef __attribute__((ext_vector_type(8)))  float  v8f;

union Frag16 {
    v16bf v;
    uint4 q[2];   // 32 bytes = 16 bf16
};

#define BATCH 64
#define TSEQ  10
#define DIN   4096
#define HID   2048
#define PRED  10

// round-to-nearest-even fp32 -> bf16 (as raw ushort)
__device__ __forceinline__ unsigned short f2bf(float f) {
    unsigned int u = __float_as_uint(f);
    u += 0x7FFFu + ((u >> 16) & 1u);
    return (unsigned short)(u >> 16);
}

__global__ __launch_bounds__(256) void cvt_f32_bf16(const float* __restrict__ in,
                                                    unsigned short* __restrict__ out,
                                                    int n) {
    int i = blockIdx.x * blockDim.x + threadIdx.x;
    int stride = gridDim.x * blockDim.x;
    for (; i < n; i += stride) out[i] = f2bf(in[i]);
}

// ---------------------------------------------------------------------------
// Fused RNN cell: Hout = tanh(A1 @ W1^T + b1 + A2 @ W2^T + b2)
// A1: [64 x K1] bf16 (row stride lda1), W1: [2048 x K1] bf16 row-major
// A2: [64 x 2048] bf16 (row stride lda2), W2: [2048 x 2048] bf16 row-major
// One wave computes one 16x16 tile of the 64x2048 output.
// Tiles: 4 (M) x 128 (N) = 512 waves = 64 blocks x 8 waves.
// ---------------------------------------------------------------------------
__global__ __launch_bounds__(256) void rnn_cell_wmma(
    const unsigned short* __restrict__ A1, int lda1, int K1,
    const unsigned short* __restrict__ W1,
    const float* __restrict__ b1,
    const unsigned short* __restrict__ A2, int lda2,
    const unsigned short* __restrict__ W2,
    const float* __restrict__ b2,
    unsigned short* __restrict__ Hout)   // [64 x 2048] bf16
{
    const int lane = threadIdx.x & 31;
    const int wave = blockIdx.x * (blockDim.x >> 5) + (threadIdx.x >> 5);
    const int m0 = (wave & 3) << 4;      // 0..48
    const int n0 = (wave >> 2) << 4;     // 0..2032

    const int lo  = lane & 15;
    const int hiA = (lane >> 4) << 4;    // +16 K offset for lanes 16..31 (A frag)
    const int hiB = (lane >> 4) << 4;    // +16 K offset for lanes 16..31 (B frag)

    v8f acc = {};

    // --- GEMM 1: A1(64xK1) * W1^T ---
    {
        const unsigned short* arow = A1 + (size_t)(m0 + lo) * lda1;
        const unsigned short* wrow = W1 + (size_t)(n0 + lo) * K1;
        for (int k = 0; k < K1; k += 32) {
            Frag16 a, b;
            a.q[0] = *(const uint4*)(arow + k + hiA);        // K = k+hiA .. +7
            a.q[1] = *(const uint4*)(arow + k + hiA + 8);    // K = k+hiA+8 .. +7
            b.q[0] = *(const uint4*)(wrow + k + hiB);        // K = k+hiB .. +7
            b.q[1] = *(const uint4*)(wrow + k + hiB + 8);    // K = k+hiB+8 .. +7
            acc = __builtin_amdgcn_wmma_f32_16x16x32_bf16(
                false, a.v, false, b.v, (short)0, acc, false, false);
        }
    }
    // --- GEMM 2: A2(64x2048) * W2^T ---
    {
        const unsigned short* arow = A2 + (size_t)(m0 + lo) * lda2;
        const unsigned short* wrow = W2 + (size_t)(n0 + lo) * HID;
        for (int k = 0; k < HID; k += 32) {
            Frag16 a, b;
            a.q[0] = *(const uint4*)(arow + k + hiA);
            a.q[1] = *(const uint4*)(arow + k + hiA + 8);
            b.q[0] = *(const uint4*)(wrow + k + hiB);
            b.q[1] = *(const uint4*)(wrow + k + hiB + 8);
            acc = __builtin_amdgcn_wmma_f32_16x16x32_bf16(
                false, a.v, false, b.v, (short)0, acc, false, false);
        }
    }

    const int n = n0 + lo;
    const float bias = b1[n] + b2[n];
    const int mbase = m0 + ((lane >> 4) << 3);   // D layout: lanes>=16 hold M+8
#pragma unroll
    for (int r = 0; r < 8; ++r) {
        float v = tanhf(acc[r] + bias);
        Hout[(size_t)(mbase + r) * HID + n] = f2bf(v);
    }
}

// ---------------------------------------------------------------------------
// Dense head: Y = act(A @ W^T + b).  A: [64 x K], W: [N x K].
// relu=1 -> ReLU; relu=0 -> identity.
// Optional bf16 output (row stride N) and optional f32 output (row stride ldf).
// ---------------------------------------------------------------------------
__global__ __launch_bounds__(256) void dense_wmma(
    const unsigned short* __restrict__ A, int lda, int K,
    const unsigned short* __restrict__ W,
    const float* __restrict__ b,
    int N, int relu,
    unsigned short* __restrict__ outB,
    float* __restrict__ outF, int ldf)
{
    const int lane = threadIdx.x & 31;
    const int wave = blockIdx.x * (blockDim.x >> 5) + (threadIdx.x >> 5);
    const int m0 = (wave & 3) << 4;
    const int nt = wave >> 2;
    if (nt >= (N >> 4)) return;
    const int n0 = nt << 4;

    const int lo  = lane & 15;
    const int hiA = (lane >> 4) << 4;
    const int hiB = (lane >> 4) << 4;

    v8f acc = {};
    const unsigned short* arow = A + (size_t)(m0 + lo) * lda;
    const unsigned short* wrow = W + (size_t)(n0 + lo) * K;
    for (int k = 0; k < K; k += 32) {
        Frag16 a, bb;
        a.q[0]  = *(const uint4*)(arow + k + hiA);
        a.q[1]  = *(const uint4*)(arow + k + hiA + 8);
        bb.q[0] = *(const uint4*)(wrow + k + hiB);
        bb.q[1] = *(const uint4*)(wrow + k + hiB + 8);
        acc = __builtin_amdgcn_wmma_f32_16x16x32_bf16(
            false, a.v, false, bb.v, (short)0, acc, false, false);
    }

    const int n = n0 + lo;
    const float bias = b[n];
    const int mbase = m0 + ((lane >> 4) << 3);
#pragma unroll
    for (int r = 0; r < 8; ++r) {
        float v = acc[r] + bias;
        if (relu) v = v > 0.0f ? v : 0.0f;
        const int m = mbase + r;
        if (outB) outB[(size_t)m * N + n] = f2bf(v);
        if (outF) outF[(size_t)m * ldf + n] = v;
    }
}

// ---------------------------------------------------------------------------
// Host-side orchestration (graph-capture safe: only kernel launches + memset)
// ---------------------------------------------------------------------------
extern "C" void kernel_launch(void* const* d_in, const int* in_sizes, int n_in,
                              void* d_out, int out_size, void* d_ws, size_t ws_size,
                              hipStream_t stream) {
    (void)in_sizes; (void)n_in; (void)out_size; (void)ws_size;

    const float* input_seq = (const float*)d_in[0];
    const float* W_ih0 = (const float*)d_in[1];
    const float* b_ih0 = (const float*)d_in[2];
    const float* W_hh0 = (const float*)d_in[3];
    const float* b_hh0 = (const float*)d_in[4];
    const float* W_ih1 = (const float*)d_in[5];
    const float* b_ih1 = (const float*)d_in[6];
    const float* W_hh1 = (const float*)d_in[7];
    const float* b_hh1 = (const float*)d_in[8];
    const float* Wd0   = (const float*)d_in[9];
    const float* bd0   = (const float*)d_in[10];
    const float* Wd1   = (const float*)d_in[11];
    const float* bd1   = (const float*)d_in[12];
    float* out = (float*)d_out;

    // workspace carve-up (256B aligned)
    char* ws = (char*)d_ws;
    size_t off = 0;
    auto take = [&](size_t bytes) -> unsigned short* {
        unsigned short* p = (unsigned short*)(ws + off);
        off = (off + bytes + 255) & ~(size_t)255;
        return p;
    };
    unsigned short* Wih0b = take((size_t)HID * DIN * 2);   // 16 MB
    unsigned short* Whh0b = take((size_t)HID * HID * 2);   //  8 MB
    unsigned short* Wih1b = take((size_t)HID * HID * 2);
    unsigned short* Whh1b = take((size_t)HID * HID * 2);
    unsigned short* Wd0b  = take((size_t)HID * HID * 2);
    unsigned short* Wd1b  = take((size_t)DIN * HID * 2);   // 16 MB
    unsigned short* Xb    = take((size_t)BATCH * TSEQ * DIN * 2);
    unsigned short* h0buf[2] = { take((size_t)BATCH * HID * 2), take((size_t)BATCH * HID * 2) };
    unsigned short* h1buf[2] = { take((size_t)BATCH * HID * 2), take((size_t)BATCH * HID * 2) };
    unsigned short* dbuf  = take((size_t)BATCH * HID * 2);
    unsigned short* prevb = take((size_t)BATCH * DIN * 2);

    // 1) convert weights + input sequence to bf16
    auto cvt = [&](const float* src, unsigned short* dst, int n) {
        cvt_f32_bf16<<<1024, 256, 0, stream>>>(src, dst, n);
    };
    cvt(W_ih0, Wih0b, HID * DIN);
    cvt(W_hh0, Whh0b, HID * HID);
    cvt(W_ih1, Wih1b, HID * HID);
    cvt(W_hh1, Whh1b, HID * HID);
    cvt(Wd0,   Wd0b,  HID * HID);
    cvt(Wd1,   Wd1b,  DIN * HID);
    cvt(input_seq, Xb, BATCH * TSEQ * DIN);

    // 2) zero initial hidden states
    hipMemsetAsync(h0buf[0], 0, (size_t)BATCH * HID * 2, stream);
    hipMemsetAsync(h1buf[0], 0, (size_t)BATCH * HID * 2, stream);

    const int CELL_BLOCKS  = (4 * (HID / 16) * 32) / 256;   // 64
    const int DENSE0_BLKS  = (4 * (HID / 16) * 32) / 256;   // 64
    const int DENSE1_BLKS  = (4 * (DIN / 16) * 32) / 256;   // 128

    int cur = 0;
    // 3) encoder: 10 steps
    for (int t = 0; t < TSEQ; ++t) {
        const unsigned short* xt = Xb + (size_t)t * DIN;     // row stride TSEQ*DIN
        int nxt = cur ^ 1;
        rnn_cell_wmma<<<CELL_BLOCKS, 256, 0, stream>>>(
            xt, TSEQ * DIN, DIN, Wih0b, b_ih0,
            h0buf[cur], HID, Whh0b, b_hh0, h0buf[nxt]);
        rnn_cell_wmma<<<CELL_BLOCKS, 256, 0, stream>>>(
            h0buf[nxt], HID, HID, Wih1b, b_ih1,
            h1buf[cur], HID, Whh1b, b_hh1, h1buf[nxt]);
        cur = nxt;
    }

    // 4) decoder: 10 steps (pred_len is a compile-time constant of the problem)
    const unsigned short* prev = Xb + (size_t)(TSEQ - 1) * DIN;  // input_seq[:, -1]
    int ldprev = TSEQ * DIN;
    for (int t = 0; t < PRED; ++t) {
        int nxt = cur ^ 1;
        rnn_cell_wmma<<<CELL_BLOCKS, 256, 0, stream>>>(
            prev, ldprev, DIN, Wih0b, b_ih0,
            h0buf[cur], HID, Whh0b, b_hh0, h0buf[nxt]);
        rnn_cell_wmma<<<CELL_BLOCKS, 256, 0, stream>>>(
            h0buf[nxt], HID, HID, Wih1b, b_ih1,
            h1buf[cur], HID, Whh1b, b_hh1, h1buf[nxt]);
        // d = relu(h1 @ Wd0^T + bd0)
        dense_wmma<<<DENSE0_BLKS, 256, 0, stream>>>(
            h1buf[nxt], HID, HID, Wd0b, bd0, HID, 1,
            dbuf, (float*)nullptr, 0);
        // out_t = d @ Wd1^T + bd1 -> d_out[b, t, :] (stride PRED*DIN), bf16 copy -> prev
        dense_wmma<<<DENSE1_BLKS, 256, 0, stream>>>(
            dbuf, HID, HID, Wd1b, bd1, DIN, 0,
            prevb, out + (size_t)t * DIN, PRED * DIN);
        prev = prevb;
        ldprev = DIN;
        cur = nxt;
    }
}
